// TransformerEncoderLayerwithRelativePE_90503550861826
// MI455X (gfx1250) — compile-verified, matrix operand order checked by reading, not run on hardware
//
#include <hip/hip_runtime.h>
#include <hip/hip_bf16.h>

// ---------------------------------------------------------------------------
// CDNA5 (gfx1250) wave32 WMMA bf16 Transformer-XL encoder layer.
// - All GEMMs: v_wmma_f32_16x16x32_bf16, 16x64 C strip per wave (4 accums).
// - B panels staged in LDS, shared by 4 waves, double-buffered with
//   GLOBAL_LOAD_ASYNC_TO_LDS (ASYNCcnt).
// - Attention: TDM tensor_load_to_lds streams the K head-panel into LDS
//   overlapped with the relative-logit WMMA phase (TENSORcnt).
// ---------------------------------------------------------------------------

typedef __attribute__((ext_vector_type(16))) __bf16 v16bf;
typedef __attribute__((ext_vector_type(8)))  float  v8f;
typedef __attribute__((ext_vector_type(4)))  unsigned int v4u;
typedef __attribute__((ext_vector_type(8)))  int    v8i;
typedef __attribute__((ext_vector_type(4)))  int    v4i;

typedef __attribute__((address_space(1))) v4i gas_v4i;  // global <4 x i32>
typedef __attribute__((address_space(3))) v4i las_v4i;  // LDS    <4 x i32>

#if defined(__has_builtin)
#if __has_builtin(__builtin_amdgcn_global_load_async_to_lds_b128) && \
    __has_builtin(__builtin_amdgcn_s_wait_asynccnt)
#define USE_ASYNC_LDS 1
#else
#define USE_ASYNC_LDS 0
#endif
#if __has_builtin(__builtin_amdgcn_tensor_load_to_lds) && \
    __has_builtin(__builtin_amdgcn_s_wait_tensorcnt)
#define USE_TDM 1
#else
#define USE_TDM 0
#endif
#else
#define USE_ASYNC_LDS 0
#define USE_TDM 0
#endif

union Frag {
    v16bf v;
    unsigned int u[8];
    __bf16 h[16];
};

static __device__ inline __bf16 f2bf(float x) {
    unsigned int u = __builtin_bit_cast(unsigned int, x);
    unsigned int r = u + 0x7fffu + ((u >> 16) & 1u);
    unsigned short s = (unsigned short)(r >> 16);
    return __builtin_bit_cast(__bf16, s);
}

static __device__ inline v8f vzero() {
    v8f z;
#pragma unroll
    for (int i = 0; i < 8; ++i) z[i] = 0.0f;
    return z;
}

static __device__ inline v8f wmma_bf16(v16bf a, v16bf b, v8f c) {
    return __builtin_amdgcn_wmma_f32_16x16x32_bf16(
        false, a, false, b, (short)0, c, false, false);
}

// 16x32 bf16 fragment from row-major (rows x K). A matrix, or B^T stored N x K.
static __device__ inline v16bf load_rk(const __bf16* p, int ld, int row0, int k0, int lane) {
    Frag f;
    const int r  = row0 + (lane & 15);
    const int hi = (lane >> 4) & 1;
    const __bf16* base = p + (size_t)r * ld + k0 + hi * 8;
#pragma unroll
    for (int v = 0; v < 8; ++v) {
        const int kb = (v & 3) * 2 + (v >> 2) * 16;
        f.u[v] = *(const unsigned int*)(base + kb);
    }
    return f.v;
}

static __device__ inline v16bf load_rk_guard(const __bf16* p, int ld, int row0, int k0,
                                             int lane, int nrows) {
    Frag f;
    const int r  = row0 + (lane & 15);
    const int hi = (lane >> 4) & 1;
    if ((unsigned)r < (unsigned)nrows) {
        const __bf16* base = p + (size_t)r * ld + k0 + hi * 8;
#pragma unroll
        for (int v = 0; v < 8; ++v) {
            const int kb = (v & 3) * 2 + (v >> 2) * 16;
            f.u[v] = *(const unsigned int*)(base + kb);
        }
    } else {
#pragma unroll
        for (int v = 0; v < 8; ++v) f.u[v] = 0u;
    }
    return f.v;
}

// 32x16 bf16 B fragment from row-major K x N (strided; used for P*V).
static __device__ inline v16bf load_kn(const __bf16* p, int ldn, int k0, int n0, int lane) {
    Frag f;
    const int n  = n0 + (lane & 15);
    const int hi = (lane >> 4) & 1;
#pragma unroll
    for (int v = 0; v < 8; ++v) {
        const int kd = k0 + (v & 3) * 2 + (v >> 2) * 16 + hi * 8;
        f.h[2 * v]     = p[(size_t)kd * ldn + n];
        f.h[2 * v + 1] = p[(size_t)(kd + 1) * ldn + n];
    }
    return f.v;
}

// ---------------------------------------------------------------------------
// Shared-panel GEMM core: 128 threads = 4 waves. Wave w computes the 16x64
// strip at (blk_m*64 + w*16, blk_n*64). Bt panel (64 rows x 32 k = 4KB)
// staged in LDS per K-step, shared by all waves; double-buffered async.
// ---------------------------------------------------------------------------
static __device__ inline void stage_panel_sync(const __bf16* Bt, int ldb, int n0, int k0,
                                               __bf16* dst, int tid) {
    unsigned int* d = (unsigned int*)dst;  // 64*32 bf16 = 1024 dwords
#pragma unroll
    for (int i = tid; i < 1024; i += 128) {
        const int r = i >> 4, c = (i & 15) * 2;
        d[i] = *(const unsigned int*)(Bt + (size_t)(n0 + r) * ldb + k0 + c);
    }
}

#if USE_ASYNC_LDS
static __device__ inline void stage_panel_async(const __bf16* Bt, int ldb, int n0, int k0,
                                                __bf16* dst, int tid) {
    // 64x32 bf16 = 256 x 16B chunks; 2 async b128 per thread
#pragma unroll
    for (int j = tid; j < 256; j += 128) {
        const int r = j >> 2, c = (j & 3) * 8;
        gas_v4i* g = (gas_v4i*)(Bt + (size_t)(n0 + r) * ldb + k0 + c);
        las_v4i* l = (las_v4i*)(dst + r * 32 + c);
        __builtin_amdgcn_global_load_async_to_lds_b128(g, l, 0, 0);
    }
}
#endif

static __device__ inline void gemm_block(const __bf16* __restrict__ A, int lda,
                                         const __bf16* __restrict__ Bt, int ldb,
                                         int K, int m0, int n0, __bf16* panel,
                                         int tid, v8f acc[4]) {
    const int lane = tid & 31;
#pragma unroll
    for (int i = 0; i < 4; ++i) acc[i] = vzero();
#if USE_ASYNC_LDS
    int buf = 0;
    stage_panel_async(Bt, ldb, n0, 0, panel, tid);
    for (int k0 = 0; k0 < K; k0 += 32) {
        if (k0 + 32 < K) {
            stage_panel_async(Bt, ldb, n0, k0 + 32, panel + (buf ^ 1) * 2048, tid);
            __builtin_amdgcn_s_wait_asynccnt(2);  // current panel arrived; next in flight
        } else {
            __builtin_amdgcn_s_wait_asynccnt(0);
        }
        __syncthreads();
        const v16bf a = load_rk(A, lda, m0, k0, lane);
        const __bf16* pb = panel + buf * 2048;
#pragma unroll
        for (int nt = 0; nt < 4; ++nt) {
            v16bf b = load_rk(pb, 32, nt * 16, 0, lane);
            acc[nt] = wmma_bf16(a, b, acc[nt]);
        }
        __syncthreads();  // all waves done reading before this buffer is re-staged
        buf ^= 1;
    }
#else
    for (int k0 = 0; k0 < K; k0 += 32) {
        __syncthreads();
        stage_panel_sync(Bt, ldb, n0, k0, panel, tid);
        __syncthreads();
        const v16bf a = load_rk(A, lda, m0, k0, lane);
#pragma unroll
        for (int nt = 0; nt < 4; ++nt) {
            v16bf b = load_rk(panel, 32, nt * 16, 0, lane);
            acc[nt] = wmma_bf16(a, b, acc[nt]);
        }
    }
#endif
}

// ---------------------------------------------------------------------------
// Conversion / setup kernels
// ---------------------------------------------------------------------------

__global__ __launch_bounds__(256)
void to_bf16_kernel(const float* __restrict__ src, __bf16* __restrict__ dst, int n) {
    int i = blockIdx.x * 256 + threadIdx.x;
    if (i < n) dst[i] = f2bf(src[i]);
}

__global__ __launch_bounds__(256)
void transpose_bf16_kernel(const float* __restrict__ src, __bf16* __restrict__ dst, int K, int N) {
    int idx = blockIdx.x * 256 + threadIdx.x;
    if (idx >= K * N) return;
    int n = idx / K;
    int k = idx - n * K;
    dst[idx] = f2bf(src[(size_t)k * N + n]);
}

__global__ __launch_bounds__(256)
void pe_kernel(__bf16* __restrict__ peb) {
    const int s = blockIdx.x;
    const float pos = (float)s - 1024.0f;
    const float ln1e4 = logf(10000.0f);
    for (int c = threadIdx.x; c < 512; c += 256) {
        int j = (c < 256) ? c : (c - 256);
        float inv = __expf(-((float)(2 * j) / 512.0f) * ln1e4);
        float a = pos * inv;
        peb[(size_t)s * 512 + c] = f2bf((c < 256) ? __sinf(a) : __cosf(a));
    }
}

// ---------------------------------------------------------------------------
// GEMM kernels (128 threads = 4 waves; 16x64 strip per wave)
// ---------------------------------------------------------------------------

__global__ __launch_bounds__(128)
void qkv_gemm_kernel(const __bf16* __restrict__ xb,
                     const __bf16* __restrict__ Wqt, const __bf16* __restrict__ Wkt,
                     const __bf16* __restrict__ Wvt,
                     const float* __restrict__ bq, const float* __restrict__ bk,
                     const float* __restrict__ bv,
                     const float* __restrict__ cbias, const float* __restrict__ rbias,
                     __bf16* __restrict__ qc, __bf16* __restrict__ qr,
                     __bf16* __restrict__ kbuf, __bf16* __restrict__ vbuf) {
    __shared__ __bf16 panel[2 * 64 * 32];
    const int tid = threadIdx.x, lane = tid & 31, wave = tid >> 5;
    const int m0 = blockIdx.x * 64 + wave * 16;
    const int n0 = blockIdx.y * 64;
    const int z  = blockIdx.z;
    const __bf16* Wt = (z == 0) ? Wqt : (z == 1) ? Wkt : Wvt;
    v8f acc[4];
    gemm_block(xb, 512, Wt, 512, 512, m0, n0, panel, tid, acc);

    const int hi = lane >> 4;
    const int nn = lane & 15;
#pragma unroll
    for (int nt = 0; nt < 4; ++nt) {
        const int n   = n0 + nt * 16 + nn;
        const int hh  = n >> 6;
        const int dhi = n & 63;
        const float bias = (z == 0) ? bq[n] : (z == 1) ? bk[n] : bv[n];
#pragma unroll
        for (int r = 0; r < 8; ++r) {
            const int m = m0 + r + 8 * hi;
            const int bi = m >> 10, t = m & 1023;
            const size_t o = (((size_t)(bi * 8 + hh)) * 1024 + t) * 64 + dhi;
            const float val = acc[nt][r] + bias;
            if (z == 0) {
                qc[o] = f2bf(val + cbias[n]);
                qr[o] = f2bf(val + rbias[n]);
            } else if (z == 1) {
                kbuf[o] = f2bf(val);
            } else {
                vbuf[o] = f2bf(val);
            }
        }
    }
}

__global__ __launch_bounds__(128)
void rel_gemm_kernel(const __bf16* __restrict__ peb, const __bf16* __restrict__ Wrt,
                     __bf16* __restrict__ rb) {
    __shared__ __bf16 panel[2 * 64 * 32];
    const int tid = threadIdx.x, lane = tid & 31, wave = tid >> 5;
    const int m0 = blockIdx.x * 64 + wave * 16;
    const int n0 = blockIdx.y * 64;
    v8f acc[4];
    gemm_block(peb, 512, Wrt, 512, 512, m0, n0, panel, tid, acc);
    const int hi = lane >> 4, nn = lane & 15;
#pragma unroll
    for (int nt = 0; nt < 4; ++nt) {
        const int n = n0 + nt * 16 + nn;
        const int hh = n >> 6, dhi = n & 63;
#pragma unroll
        for (int r = 0; r < 8; ++r) {
            const int s = m0 + r + 8 * hi;
            rb[((size_t)hh * 2048 + s) * 64 + dhi] = f2bf(acc[nt][r]);
        }
    }
}

__global__ __launch_bounds__(128)
void gemm_bias_f32_kernel(const __bf16* __restrict__ A, const __bf16* __restrict__ Bt,
                          const float* __restrict__ bias, float* __restrict__ C,
                          int K, int N) {
    __shared__ __bf16 panel[2 * 64 * 32];
    const int tid = threadIdx.x, lane = tid & 31, wave = tid >> 5;
    const int m0 = blockIdx.x * 64 + wave * 16;
    const int n0 = blockIdx.y * 64;
    v8f acc[4];
    gemm_block(A, K, Bt, K, K, m0, n0, panel, tid, acc);
    const int hi = lane >> 4, nn = lane & 15;
#pragma unroll
    for (int nt = 0; nt < 4; ++nt) {
        const int n = n0 + nt * 16 + nn;
#pragma unroll
        for (int r = 0; r < 8; ++r) {
            const int m = m0 + r + 8 * hi;
            C[(size_t)m * N + n] = acc[nt][r] + bias[n];
        }
    }
}

__global__ __launch_bounds__(128)
void gemm_bias_relu_bf16_kernel(const __bf16* __restrict__ A, const __bf16* __restrict__ Bt,
                                const float* __restrict__ bias, __bf16* __restrict__ C,
                                int K, int N) {
    __shared__ __bf16 panel[2 * 64 * 32];
    const int tid = threadIdx.x, lane = tid & 31, wave = tid >> 5;
    const int m0 = blockIdx.x * 64 + wave * 16;
    const int n0 = blockIdx.y * 64;
    v8f acc[4];
    gemm_block(A, K, Bt, K, K, m0, n0, panel, tid, acc);
    const int hi = lane >> 4, nn = lane & 15;
#pragma unroll
    for (int nt = 0; nt < 4; ++nt) {
        const int n = n0 + nt * 16 + nn;
#pragma unroll
        for (int r = 0; r < 8; ++r) {
            const int m = m0 + r + 8 * hi;
            C[(size_t)m * N + n] = f2bf(fmaxf(acc[nt][r] + bias[n], 0.0f));
        }
    }
}

// ---------------------------------------------------------------------------
// Attention kernel: one wave per (b, h, 16-row query tile).
// logits[t, j] = qc[t].k[j] + qr[t].rel[1024 + j - t]
// TDM streams the K head-panel (1024x64 bf16 = 128KB) into LDS while the
// wave computes the relative-logit window; s_wait_tensorcnt gates phase 2.
// LDS: K panel 131072 | RL 16x1056 f32 (67584) | LG 16x1024 f32 (65536)
// ---------------------------------------------------------------------------
#define ATTN_KL_BYTES   (1024 * 64 * 2)
#define ATTN_RL_OFF     ATTN_KL_BYTES
#define ATTN_LG_OFF     (ATTN_RL_OFF + 16 * 1056 * 4)
#define ATTN_LDS_BYTES  (ATTN_LG_OFF + 16 * 1024 * 4)

__global__ __launch_bounds__(32)
void attn_kernel(const __bf16* __restrict__ qc, const __bf16* __restrict__ qr,
                 const __bf16* __restrict__ kb, const __bf16* __restrict__ vb,
                 const __bf16* __restrict__ rb, __bf16* __restrict__ ob) {
    extern __shared__ char smem[];
    float*  RL = (float*)(smem + ATTN_RL_OFF);   // [16][1056]
    float*  LG = (float*)(smem + ATTN_LG_OFF);   // [16][1024]
    __bf16* P  = (__bf16*)(smem + ATTN_RL_OFF);  // [16][1024] (reuses RL)

    const int lane = threadIdx.x;
    const int t0 = blockIdx.x * 16;
    const int h  = blockIdx.y;
    const int bi = blockIdx.z;
    const size_t hb = ((size_t)(bi * 8 + h)) * 1024 * 64;
    const __bf16* qch = qc + hb;
    const __bf16* qrh = qr + hb;
    const __bf16* kbh = kb + hb;
    const __bf16* vbh = vb + hb;
    const __bf16* rbh = rb + (size_t)h * 2048 * 64;

#if USE_TDM
    // Issue TDM: 2D tile 64 x 1024 (elem 2B) from kbh into LDS offset 0.
    {
        const unsigned long long ga = (unsigned long long)kbh;
        v4u g0;
        g0[0] = 1u;                                    // count=1
        g0[1] = 0u;                                    // lds_addr
        g0[2] = (unsigned int)(ga & 0xffffffffu);      // global_addr[31:0]
        g0[3] = (unsigned int)((ga >> 32) & 0x1ffffffu) | (2u << 30);  // addr[56:32], type=2
        v8i g1;
        g1[0] = (int)(1u << 16);     // data_size = 1 (2 bytes)
        g1[1] = (int)(64u << 16);    // tensor_dim0 = 64
        g1[2] = (int)(1024u << 16);  // tensor_dim1 = 1024
        g1[3] = (int)(64u << 16);    // tile_dim0 = 64
        g1[4] = 1024;                // tile_dim1 = 1024
        g1[5] = 64;                  // tensor_dim0_stride = 64
        g1[6] = 0;
        g1[7] = 0;
        v4i g2, g3;
#pragma unroll
        for (int i = 0; i < 4; ++i) { g2[i] = 0; g3[i] = 0; }
        v8i g4;
#pragma unroll
        for (int i = 0; i < 8; ++i) g4[i] = 0;
        __builtin_amdgcn_tensor_load_to_lds(g0, g1, g2, g3, g4, 0);
    }
    const __bf16* ksrc = (const __bf16*)smem;   // K panel now streaming into LDS
#else
    const __bf16* ksrc = kbh;                   // fallback: read K from global/L2
#endif

    const int hi = lane >> 4;
    const int nn = lane & 15;

    v16bf aqc0 = load_rk(qch, 64, t0, 0, lane);
    v16bf aqc1 = load_rk(qch, 64, t0, 32, lane);
    v16bf aqr0 = load_rk(qrh, 64, t0, 0, lane);
    v16bf aqr1 = load_rk(qrh, 64, t0, 32, lane);

    // Phase 1 (overlapped with TDM): RL[m][w] = qr[t0+m] . rel[w0+w]
    const int w0 = 1024 - t0 - 16;
    for (int wt = 0; wt < 66; ++wt) {
        v16bf b0 = load_rk_guard(rbh, 64, w0 + wt * 16, 0, lane, 2048);
        v16bf b1 = load_rk_guard(rbh, 64, w0 + wt * 16, 32, lane, 2048);
        v8f acc = vzero();
        acc = wmma_bf16(aqr0, b0, acc);
        acc = wmma_bf16(aqr1, b1, acc);
#pragma unroll
        for (int r = 0; r < 8; ++r)
            RL[(r + 8 * hi) * 1056 + wt * 16 + nn] = acc[r];
    }
    __syncthreads();
#if USE_TDM
    __builtin_amdgcn_s_wait_tensorcnt(0);   // K panel resident in LDS
#endif

    // Phase 2: content logits + diagonal gather of RL
    for (int jt = 0; jt < 64; ++jt) {
        v16bf b0 = load_rk(ksrc, 64, jt * 16, 0, lane);
        v16bf b1 = load_rk(ksrc, 64, jt * 16, 32, lane);
        v8f acc = vzero();
        acc = wmma_bf16(aqc0, b0, acc);
        acc = wmma_bf16(aqc1, b1, acc);
#pragma unroll
        for (int r = 0; r < 8; ++r) {
            const int m = r + 8 * hi;
            LG[m * 1024 + jt * 16 + nn] = acc[r] + RL[m * 1056 + jt * 16 + nn - m + 16];
        }
    }
    __syncthreads();

    // Phase 3: two-pass softmax; lane = (row nn, half hi)
    {
        const float* row = LG + nn * 1024 + hi * 512;
        float mx = -1e30f;
        for (int c = 0; c < 512; ++c) mx = fmaxf(mx, row[c]);
        mx = fmaxf(mx, __shfl_xor(mx, 16, 32));
        float sum = 0.0f;
        for (int c = 0; c < 512; ++c) sum += __expf(row[c] - mx);
        sum += __shfl_xor(sum, 16, 32);
        const float rinv = 1.0f / sum;
        __bf16* prow = P + nn * 1024 + hi * 512;
        for (int c = 0; c < 512; ++c)
            prow[c] = f2bf(__expf(row[c] - mx) * rinv);
    }
    __syncthreads();

    // Phase 4: O(16x64) = P(16x1024) * V(1024x64)
    v8f oacc[4];
#pragma unroll
    for (int nt = 0; nt < 4; ++nt) oacc[nt] = vzero();
    for (int kt = 0; kt < 32; ++kt) {
        v16bf a = load_rk((const __bf16*)P, 1024, 0, kt * 32, lane);
#pragma unroll
        for (int nt = 0; nt < 4; ++nt) {
            v16bf b = load_kn(vbh, 64, kt * 32, nt * 16, lane);
            oacc[nt] = wmma_bf16(a, b, oacc[nt]);
        }
    }
#pragma unroll
    for (int nt = 0; nt < 4; ++nt) {
#pragma unroll
        for (int r = 0; r < 8; ++r) {
            const int m = r + 8 * hi;
            ob[(size_t)(bi * 1024 + t0 + m) * 512 + h * 64 + nt * 16 + nn] = f2bf(oacc[nt][r]);
        }
    }
}

// ---------------------------------------------------------------------------
// LayerNorm over rows of 512 ("x + x" doubling folded in).
// ---------------------------------------------------------------------------
__global__ __launch_bounds__(256)
void ln_kernel(const float* __restrict__ in, const float* __restrict__ scale,
               const float* __restrict__ bias, float* __restrict__ outf,
               __bf16* __restrict__ outb) {
    __shared__ float red[256];
    const int row = blockIdx.x;
    const int tid = threadIdx.x;
    const float* r = in + (size_t)row * 512;
    const float v0 = 2.0f * r[tid];
    const float v1 = 2.0f * r[tid + 256];
    red[tid] = v0 + v1;
    __syncthreads();
    for (int o = 128; o > 0; o >>= 1) {
        if (tid < o) red[tid] += red[tid + o];
        __syncthreads();
    }
    const float mu = red[0] * (1.0f / 512.0f);
    __syncthreads();
    const float d0 = v0 - mu, d1 = v1 - mu;
    red[tid] = d0 * d0 + d1 * d1;
    __syncthreads();
    for (int o = 128; o > 0; o >>= 1) {
        if (tid < o) red[tid] += red[tid + o];
        __syncthreads();
    }
    const float rs = rsqrtf(red[0] * (1.0f / 512.0f) + 1e-6f);
    const float y0 = d0 * rs * scale[tid] + bias[tid];
    const float y1 = d1 * rs * scale[tid + 256] + bias[tid + 256];
    if (outf) {
        outf[(size_t)row * 512 + tid] = y0;
        outf[(size_t)row * 512 + tid + 256] = y1;
    }
    if (outb) {
        outb[(size_t)row * 512 + tid] = f2bf(y0);
        outb[(size_t)row * 512 + tid + 256] = f2bf(y1);
    }
}

// ---------------------------------------------------------------------------
// Host launcher
// ---------------------------------------------------------------------------
extern "C" void kernel_launch(void* const* d_in, const int* in_sizes, int n_in,
                              void* d_out, int out_size, void* d_ws, size_t ws_size,
                              hipStream_t stream) {
    const float* x   = (const float*)d_in[0];
    const float* Wq  = (const float*)d_in[1];
    const float* bq  = (const float*)d_in[2];
    const float* Wk  = (const float*)d_in[3];
    const float* bk  = (const float*)d_in[4];
    const float* Wv  = (const float*)d_in[5];
    const float* bv  = (const float*)d_in[6];
    const float* cb  = (const float*)d_in[7];
    const float* rbb = (const float*)d_in[8];
    const float* Wr  = (const float*)d_in[9];
    const float* Wo  = (const float*)d_in[10];
    const float* bo  = (const float*)d_in[11];
    const float* l1s = (const float*)d_in[12];
    const float* l1b = (const float*)d_in[13];
    const float* W1  = (const float*)d_in[14];
    const float* b1  = (const float*)d_in[15];
    const float* W2  = (const float*)d_in[16];
    const float* b2  = (const float*)d_in[17];
    const float* l2s = (const float*)d_in[18];
    const float* l2b = (const float*)d_in[19];

    size_t off = 0;
    auto carve = [&](size_t bytes) -> void* {
        void* p = (char*)d_ws + off;
        off += (bytes + 255) & ~(size_t)255;
        return p;
    };
    __bf16* xb  = (__bf16*)carve((size_t)4096 * 512 * 2);
    __bf16* peb = (__bf16*)carve((size_t)2048 * 512 * 2);
    __bf16* Wqt = (__bf16*)carve((size_t)512 * 512 * 2);
    __bf16* Wkt = (__bf16*)carve((size_t)512 * 512 * 2);
    __bf16* Wvt = (__bf16*)carve((size_t)512 * 512 * 2);
    __bf16* Wrt = (__bf16*)carve((size_t)512 * 512 * 2);
    __bf16* Wot = (__bf16*)carve((size_t)512 * 512 * 2);
    __bf16* W1t = (__bf16*)carve((size_t)2048 * 512 * 2);
    __bf16* W2t = (__bf16*)carve((size_t)512 * 2048 * 2);
    __bf16* qcb = (__bf16*)carve((size_t)4 * 8 * 1024 * 64 * 2);
    __bf16* qrb = (__bf16*)carve((size_t)4 * 8 * 1024 * 64 * 2);
    __bf16* kbb = (__bf16*)carve((size_t)4 * 8 * 1024 * 64 * 2);
    __bf16* vbb = (__bf16*)carve((size_t)4 * 8 * 1024 * 64 * 2);
    __bf16* rb  = (__bf16*)carve((size_t)8 * 2048 * 64 * 2);
    __bf16* ob  = (__bf16*)carve((size_t)4096 * 512 * 2);
    float*  o1  = (float*)carve((size_t)4096 * 512 * 4);
    __bf16* y1b = (__bf16*)carve((size_t)4096 * 512 * 2);
    __bf16* f1b = (__bf16*)carve((size_t)4096 * 2048 * 2);
    float*  f2  = (float*)carve((size_t)4096 * 512 * 4);

    to_bf16_kernel<<<(4096 * 512 + 255) / 256, 256, 0, stream>>>(x, xb, 4096 * 512);
    transpose_bf16_kernel<<<(512 * 512 + 255) / 256, 256, 0, stream>>>(Wq, Wqt, 512, 512);
    transpose_bf16_kernel<<<(512 * 512 + 255) / 256, 256, 0, stream>>>(Wk, Wkt, 512, 512);
    transpose_bf16_kernel<<<(512 * 512 + 255) / 256, 256, 0, stream>>>(Wv, Wvt, 512, 512);
    transpose_bf16_kernel<<<(512 * 512 + 255) / 256, 256, 0, stream>>>(Wr, Wrt, 512, 512);
    transpose_bf16_kernel<<<(512 * 512 + 255) / 256, 256, 0, stream>>>(Wo, Wot, 512, 512);
    transpose_bf16_kernel<<<(512 * 2048 + 255) / 256, 256, 0, stream>>>(W1, W1t, 512, 2048);
    transpose_bf16_kernel<<<(2048 * 512 + 255) / 256, 256, 0, stream>>>(W2, W2t, 2048, 512);
    pe_kernel<<<2048, 256, 0, stream>>>(peb);

    qkv_gemm_kernel<<<dim3(64, 8, 3), 128, 0, stream>>>(
        xb, Wqt, Wkt, Wvt, bq, bk, bv, cb, rbb, qcb, qrb, kbb, vbb);
    rel_gemm_kernel<<<dim3(32, 8), 128, 0, stream>>>(peb, Wrt, rb);

    attn_kernel<<<dim3(64, 8, 4), 32, ATTN_LDS_BYTES, stream>>>(qcb, qrb, kbb, vbb, rb, ob);

    gemm_bias_f32_kernel<<<dim3(64, 8), 128, 0, stream>>>(ob, Wot, bo, o1, 512, 512);
    ln_kernel<<<4096, 256, 0, stream>>>(o1, l1s, l1b, nullptr, y1b);

    gemm_bias_relu_bf16_kernel<<<dim3(64, 32), 128, 0, stream>>>(y1b, W1t, b1, f1b, 512, 2048);
    gemm_bias_f32_kernel<<<dim3(64, 8), 128, 0, stream>>>(f1b, W2t, b2, f2, 2048, 512);
    ln_kernel<<<4096, 256, 0, stream>>>(f2, l2s, l2b, (float*)d_out, nullptr);
}
